// enhance_net_nopool_2731599200836
// MI455X (gfx1250) — matrix operand order repaired and checked by reading, not compiled, not run on hardware
//
#include <hip/hip_runtime.h>

// ---------------------------------------------------------------------------
// enhance-net for MI455X (gfx1250, wave32)
//  - 7 fused depthwise+pointwise conv stages, f32 math, f16 feature storage
//    (halves HBM traffic; producer+consumer buffers fit 192MB L2)
//  - tiny transformer branch: 1 workgroup/image, all-LDS, GEMMs via
//    v_wmma_f32_16x16x32_f16; weights + activations staged in LDS as
//    zero-padded f16 so fragment loads are unconditional ds_load_b128s
//  - final per-pixel enhancement math
// ---------------------------------------------------------------------------

#define HW    (1024*1024)
#define NPIX  (8*1024*1024)

// d_out float offsets: (enhance, enhance, r, g, b, t)
#define E0OFF ((size_t)0)
#define E1OFF ((size_t)25165824)
#define ROFF  ((size_t)50331648)
#define GOFF  ((size_t)58720256)
#define BOFF  ((size_t)58720257)
#define TOFF  ((size_t)58720258)

typedef _Float16 h4f  __attribute__((ext_vector_type(4)));
typedef _Float16 v16h __attribute__((ext_vector_type(16)));
typedef float    v8f  __attribute__((ext_vector_type(8)));

// ---------------- luminance ----------------
__global__ void k_lum(const float* __restrict__ X, float* __restrict__ V) {
  int idx = blockIdx.x * 256 + threadIdx.x;
  if (idx >= NPIX) return;
  int b = idx >> 20; int rem = idx & (HW - 1);
  size_t xb = (size_t)b * 3 * HW + rem;
  V[idx] = (X[xb] + X[xb + HW] + X[xb + 2 * HW]) * (1.0f / 3.0f);
}

// ---------------- stage 1: 3x3 (1->1) + pw (1->4) + relu ----------------
__global__ void k_stage1(const float* __restrict__ V, h4f* __restrict__ O,
                         const float* __restrict__ ew, const float* __restrict__ eb,
                         const float* __restrict__ pw, const float* __restrict__ pb) {
  int idx = blockIdx.x * 256 + threadIdx.x;
  if (idx >= NPIX) return;
  int x = idx & 1023, y = (idx >> 10) & 1023;
  int base = idx - (y << 10) - x;
  float s = eb[0];
#pragma unroll
  for (int ky = 0; ky < 3; ky++) {
    int yy = y + ky - 1; if ((unsigned)yy > 1023u) continue;
#pragma unroll
    for (int kx = 0; kx < 3; kx++) {
      int xx = x + kx - 1; if ((unsigned)xx > 1023u) continue;
      s += V[base + (yy << 10) + xx] * ew[ky * 3 + kx];
    }
  }
  h4f o;
#pragma unroll
  for (int c = 0; c < 4; c++) {
    float t = s * pw[c] + pb[c];
    o[c] = (_Float16)fmaxf(t, 0.f);
  }
  O[idx] = o;
}

// ---------------- depthwise 3x3 (4ch) + pw (4->4) + relu ----------------
__global__ void k_dwpw4(const h4f* __restrict__ I, h4f* __restrict__ O,
                        const float* __restrict__ ew, const float* __restrict__ eb,
                        const float* __restrict__ pw, const float* __restrict__ pb) {
  int idx = blockIdx.x * 256 + threadIdx.x;
  if (idx >= NPIX) return;
  int x = idx & 1023, y = (idx >> 10) & 1023;
  int base = idx - (y << 10) - x;
  float a[4] = {eb[0], eb[1], eb[2], eb[3]};
#pragma unroll
  for (int ky = 0; ky < 3; ky++) {
    int yy = y + ky - 1; if ((unsigned)yy > 1023u) continue;
#pragma unroll
    for (int kx = 0; kx < 3; kx++) {
      int xx = x + kx - 1; if ((unsigned)xx > 1023u) continue;
      h4f t = I[base + (yy << 10) + xx];
      int tap = ky * 3 + kx;
#pragma unroll
      for (int c = 0; c < 4; c++) a[c] += (float)t[c] * ew[c * 9 + tap];
    }
  }
  h4f o;
#pragma unroll
  for (int j = 0; j < 4; j++) {
    float s = pb[j];
#pragma unroll
    for (int c = 0; c < 4; c++) s += pw[j * 4 + c] * a[c];
    o[j] = (_Float16)fmaxf(s, 0.f);
  }
  O[idx] = o;
}

// -------- depthwise 3x3 (8ch = concat(A,B)) + pw (8->4) + relu --------
__global__ void k_dwpw8(const h4f* __restrict__ IA, const h4f* __restrict__ IB,
                        h4f* __restrict__ O,
                        const float* __restrict__ ew, const float* __restrict__ eb,
                        const float* __restrict__ pw, const float* __restrict__ pb) {
  int idx = blockIdx.x * 256 + threadIdx.x;
  if (idx >= NPIX) return;
  int x = idx & 1023, y = (idx >> 10) & 1023;
  int base = idx - (y << 10) - x;
  float a[8];
#pragma unroll
  for (int c = 0; c < 8; c++) a[c] = eb[c];
#pragma unroll
  for (int ky = 0; ky < 3; ky++) {
    int yy = y + ky - 1; if ((unsigned)yy > 1023u) continue;
#pragma unroll
    for (int kx = 0; kx < 3; kx++) {
      int xx = x + kx - 1; if ((unsigned)xx > 1023u) continue;
      int off = base + (yy << 10) + xx;
      h4f ta = IA[off], tb = IB[off];
      int tap = ky * 3 + kx;
#pragma unroll
      for (int c = 0; c < 4; c++) {
        a[c]     += (float)ta[c] * ew[c * 9 + tap];
        a[4 + c] += (float)tb[c] * ew[(4 + c) * 9 + tap];
      }
    }
  }
  h4f o;
#pragma unroll
  for (int j = 0; j < 4; j++) {
    float s = pb[j];
#pragma unroll
    for (int c = 0; c < 8; c++) s += pw[j * 8 + c] * a[c];
    o[j] = (_Float16)fmaxf(s, 0.f);
  }
  O[idx] = o;
}

// -------- stage 7: depthwise 3x3 (8ch) + pw (8->1) + sigmoid -> r --------
__global__ void k_stage7(const h4f* __restrict__ IA, const h4f* __restrict__ IB,
                         float* __restrict__ R,
                         const float* __restrict__ ew, const float* __restrict__ eb,
                         const float* __restrict__ pw, const float* __restrict__ pb) {
  int idx = blockIdx.x * 256 + threadIdx.x;
  if (idx >= NPIX) return;
  int x = idx & 1023, y = (idx >> 10) & 1023;
  int base = idx - (y << 10) - x;
  float a[8];
#pragma unroll
  for (int c = 0; c < 8; c++) a[c] = eb[c];
#pragma unroll
  for (int ky = 0; ky < 3; ky++) {
    int yy = y + ky - 1; if ((unsigned)yy > 1023u) continue;
#pragma unroll
    for (int kx = 0; kx < 3; kx++) {
      int xx = x + kx - 1; if ((unsigned)xx > 1023u) continue;
      int off = base + (yy << 10) + xx;
      h4f ta = IA[off], tb = IB[off];
      int tap = ky * 3 + kx;
#pragma unroll
      for (int c = 0; c < 4; c++) {
        a[c]     += (float)ta[c] * ew[c * 9 + tap];
        a[4 + c] += (float)tb[c] * ew[(4 + c) * 9 + tap];
      }
    }
  }
  float s = pb[0];
#pragma unroll
  for (int c = 0; c < 8; c++) s += pw[c] * a[c];
  R[idx] = 1.f / (1.f + __expf(-s));
}

// ---------------- WMMA fragment helpers (CDNA5 ISA 7.12.2 layouts) -----
// A fragment from f16 LDS, row-major, ld halfs per row, K tile at k0.
// Per lane: two contiguous 8-half runs -> 2x ds_load_b128.
__device__ inline v16h ldA_h(const _Float16* S, int ld, int m0, int k0, int lane) {
  int m = m0 + (lane & 15); int hi = (lane >> 4) & 1;
  const _Float16* p = S + m * ld + k0 + hi * 8;
  v16h a;
#pragma unroll
  for (int i = 0; i < 8; i++) { a[i] = p[i]; a[i + 8] = p[16 + i]; }
  return a;
}
// B fragment from f16 LDS weight tile W[n][k] (out = in @ W^T), ld halfs/row.
// Per lane: two contiguous 8-half runs -> 2x ds_load_b128.
__device__ inline v16h ldB_h(const _Float16* Wh, int ld, int n0, int k0, int lane) {
  int n = n0 + (lane & 15); int hi = (lane >> 4) & 1;
  const _Float16* p = Wh + n * ld + k0 + hi * 8;
  v16h b;
#pragma unroll
  for (int i = 0; i < 8; i++) { b[i] = p[i]; b[i + 8] = p[16 + i]; }
  return b;
}
__device__ inline v8f wmma_f16(v16h a, v16h b, v8f c) {
  return __builtin_amdgcn_wmma_f32_16x16x32_f16(false, a, false, b, (short)0, c,
                                                false, false);
}

// ---------------- small branch: conv m + IN + transformer + level ------
__global__ void __launch_bounds__(256) k_small(
    const float* __restrict__ V,
    const float* __restrict__ mw, const float* __restrict__ mb,
    const float* __restrict__ ing, const float* __restrict__ inb,
    const float* __restrict__ aiw, const float* __restrict__ aib,
    const float* __restrict__ aow, const float* __restrict__ aob,
    const float* __restrict__ l1w, const float* __restrict__ l1b,
    const float* __restrict__ l2w, const float* __restrict__ l2b,
    const float* __restrict__ n1g, const float* __restrict__ n1b,
    const float* __restrict__ n2g, const float* __restrict__ n2b,
    const float* __restrict__ Fw, const float* __restrict__ Fb,
    float* __restrict__ GB, float* __restrict__ BBv, float* __restrict__ OUT) {
  extern __shared__ char smem[];
  // f32 region
  float* QKV  = (float*)smem;      // 256*48
  float* TMP  = QKV + 12288;       // 256*16
  float* SRCf = TMP + 4096;        // 256*16
  float* V32  = SRCf + 4096;       // 1024
  float* HB   = V32 + 1024;        // 16*256
  float* RED  = HB + 4096;         // 256
  float* ST   = RED + 256;         // 32
  // f16 region (activations padded K 16->32; weights padded likewise)
  _Float16* SRCh  = (_Float16*)(ST + 32);  // 256*32
  _Float16* OBh   = SRCh + 256 * 32;       // 256*32
  _Float16* H1h   = OBh + 256 * 32;        // 256*128
  _Float16* WQKVh = H1h + 256 * 128;       // 48*32
  _Float16* WOh   = WQKVh + 48 * 32;       // 16*32
  _Float16* W1h   = WOh + 16 * 32;         // 128*32
  _Float16* W2h   = W1h + 128 * 32;        // 16*128

  const int b = blockIdx.x;
  const int tid = threadIdx.x;
  const int lane = tid & 31;
  const int wave = __builtin_amdgcn_readfirstlane(tid >> 5);  // scalar

  // ---- stage weights into padded f16 LDS tiles ----
  for (int i = tid; i < 48 * 32; i += 256) {
    int r = i >> 5, k = i & 31;
    WQKVh[i] = (k < 16) ? (_Float16)aiw[r * 16 + k] : (_Float16)0.f;
  }
  for (int i = tid; i < 16 * 32; i += 256) {
    int r = i >> 5, k = i & 31;
    WOh[i] = (k < 16) ? (_Float16)aow[r * 16 + k] : (_Float16)0.f;
  }
  for (int i = tid; i < 128 * 32; i += 256) {
    int r = i >> 5, k = i & 31;
    W1h[i] = (k < 16) ? (_Float16)l1w[r * 16 + k] : (_Float16)0.f;
  }
  for (int i = tid; i < 16 * 128; i += 256) W2h[i] = (_Float16)l2w[i];
  // zero pad columns of activation mirrors (never rewritten)
#pragma unroll
  for (int k = 16; k < 32; k++) {
    SRCh[tid * 32 + k] = (_Float16)0.f;
    OBh[tid * 32 + k]  = (_Float16)0.f;
  }

  // ---- v32 gather (every 32nd row/col of v) ----
  for (int i = tid; i < 1024; i += 256) {
    int iy = (i >> 5) << 5;
    int ix = (i & 31) << 5;
    V32[i] = V[(size_t)b * HW + iy * 1024 + ix];
  }
  __syncthreads();

  // ---- conv m: 3x3 stride2 pad1 (1->16) + leaky(0.2), store [c][s] ----
  {
    int oy = tid >> 4, ox = tid & 15;
    float h[16];
#pragma unroll
    for (int c = 0; c < 16; c++) h[c] = mb[c];
#pragma unroll
    for (int ky = 0; ky < 3; ky++) {
      int iy = 2 * oy - 1 + ky; if ((unsigned)iy > 31u) continue;
#pragma unroll
      for (int kx = 0; kx < 3; kx++) {
        int ix = 2 * ox - 1 + kx; if ((unsigned)ix > 31u) continue;
        float vv = V32[iy * 32 + ix];
#pragma unroll
        for (int c = 0; c < 16; c++) h[c] += vv * mw[c * 9 + ky * 3 + kx];
      }
    }
#pragma unroll
    for (int c = 0; c < 16; c++) {
      float x = h[c];
      HB[c * 256 + tid] = x > 0.f ? x : 0.2f * x;
    }
  }
  __syncthreads();
  // instance-norm stats per channel
  if (tid < 16) {
    float s = 0.f, s2 = 0.f;
    for (int i = 0; i < 256; i++) { float x = HB[tid * 256 + i]; s += x; s2 += x * x; }
    float mu = s * (1.f / 256.f);
    float var = s2 * (1.f / 256.f) - mu * mu;
    ST[tid] = mu; ST[16 + tid] = rsqrtf(var + 1e-5f);
  }
  __syncthreads();
#pragma unroll
  for (int e = 0; e < 16; e++) {
    float x = (HB[e * 256 + tid] - ST[e]) * ST[16 + e] * ing[e] + inb[e];
    SRCf[tid * 16 + e] = x;
    SRCh[tid * 32 + e] = (_Float16)x;
  }
  __syncthreads();

  // ---- qkv = src @ Wqkv^T + b  (M=256,N=48,K=16) via WMMA ----
  for (int job = wave; job < 48; job += 8) {
    int mt = job & 15, nt = job >> 4;
    v16h a = ldA_h(SRCh, 32, mt * 16, 0, lane);
    v16h bf = ldB_h(WQKVh, 32, nt * 16, 0, lane);
    v8f c = {};
    c = wmma_f16(a, bf, c);
    int n = nt * 16 + (lane & 15);
    int hi = (lane >> 4) & 1;
#pragma unroll
    for (int j = 0; j < 8; j++) QKV[(mt * 16 + j + hi * 8) * 48 + n] = c[j] + aib[n];
  }
  __syncthreads();

  // ---- attention: 8 heads, Dh=2, online softmax, one row per thread ----
  {
    const float scale = 0.70710678118f;  // 1/sqrt(2)
    int s = tid;
#pragma unroll 1
    for (int h = 0; h < 8; h++) {
      float q0 = QKV[s * 48 + 2 * h], q1 = QKV[s * 48 + 2 * h + 1];
      float mmax = -3.4e38f, lsum = 0.f, a0 = 0.f, a1 = 0.f;
      for (int t = 0; t < 256; t++) {
        float k0 = QKV[t * 48 + 16 + 2 * h], k1 = QKV[t * 48 + 17 + 2 * h];
        float logit = (q0 * k0 + q1 * k1) * scale;
        float nm = fmaxf(mmax, logit);
        float corr = __expf(mmax - nm);
        float p = __expf(logit - nm);
        lsum = lsum * corr + p;
        a0 = a0 * corr + p * QKV[t * 48 + 32 + 2 * h];
        a1 = a1 * corr + p * QKV[t * 48 + 33 + 2 * h];
        mmax = nm;
      }
      float inv = 1.f / lsum;
      OBh[s * 32 + 2 * h]     = (_Float16)(a0 * inv);
      OBh[s * 32 + 2 * h + 1] = (_Float16)(a1 * inv);
    }
  }
  __syncthreads();

  // ---- attn-out projection (M=256,N=16,K=16) via WMMA ----
  for (int job = wave; job < 16; job += 8) {
    int mt = job;
    v16h a = ldA_h(OBh, 32, mt * 16, 0, lane);
    v16h bf = ldB_h(WOh, 32, 0, 0, lane);
    v8f c = {};
    c = wmma_f16(a, bf, c);
    int n = lane & 15; int hi = (lane >> 4) & 1;
#pragma unroll
    for (int j = 0; j < 8; j++) TMP[(mt * 16 + j + hi * 8) * 16 + n] = c[j] + aob[n];
  }
  __syncthreads();
  // residual + LN1 -> SRCf / SRCh
  {
    float row[16]; float s = 0.f;
#pragma unroll
    for (int e = 0; e < 16; e++) { row[e] = SRCf[tid * 16 + e] + TMP[tid * 16 + e]; s += row[e]; }
    float mu = s * (1.f / 16.f); float s2 = 0.f;
#pragma unroll
    for (int e = 0; e < 16; e++) { float d = row[e] - mu; s2 += d * d; }
    float rstd = rsqrtf(s2 * (1.f / 16.f) + 1e-5f);
    __syncthreads();
#pragma unroll
    for (int e = 0; e < 16; e++) {
      float x = (row[e] - mu) * rstd * n1g[e] + n1b[e];
      SRCf[tid * 16 + e] = x;
      SRCh[tid * 32 + e] = (_Float16)x;
    }
  }
  __syncthreads();

  // ---- FF1: relu(src @ l1^T + b1)  (M=256,N=128,K=16), store f16 ----
  for (int job = wave; job < 128; job += 8) {
    int mt = job & 15, nt = job >> 4;
    v16h a = ldA_h(SRCh, 32, mt * 16, 0, lane);
    v16h bf = ldB_h(W1h, 32, nt * 16, 0, lane);
    v8f c = {};
    c = wmma_f16(a, bf, c);
    int n = nt * 16 + (lane & 15); int hi = (lane >> 4) & 1;
#pragma unroll
    for (int j = 0; j < 8; j++) {
      float x = c[j] + l1b[n];
      H1h[(mt * 16 + j + hi * 8) * 128 + n] = (_Float16)fmaxf(x, 0.f);
    }
  }
  __syncthreads();

  // ---- FF2: H1 @ l2^T + b2  (M=256,N=16,K=128 = 4 chained K-tiles) ----
  for (int job = wave; job < 16; job += 8) {
    int mt = job;
    v8f c = {};
#pragma unroll
    for (int kt = 0; kt < 4; kt++) {
      v16h a = ldA_h(H1h, 128, mt * 16, kt * 32, lane);
      v16h bf = ldB_h(W2h, 128, 0, kt * 32, lane);
      c = wmma_f16(a, bf, c);
    }
    int n = lane & 15; int hi = (lane >> 4) & 1;
#pragma unroll
    for (int j = 0; j < 8; j++) TMP[(mt * 16 + j + hi * 8) * 16 + n] = c[j] + l2b[n];
  }
  __syncthreads();
  // residual + LN2 -> V5 (into SRCf)
  {
    float row[16]; float s = 0.f;
#pragma unroll
    for (int e = 0; e < 16; e++) { row[e] = SRCf[tid * 16 + e] + TMP[tid * 16 + e]; s += row[e]; }
    float mu = s * (1.f / 16.f); float s2 = 0.f;
#pragma unroll
    for (int e = 0; e < 16; e++) { float d = row[e] - mu; s2 += d * d; }
    float rstd = rsqrtf(s2 * (1.f / 16.f) + 1e-5f);
    __syncthreads();
#pragma unroll
    for (int e = 0; e < 16; e++) SRCf[tid * 16 + e] = (row[e] - mu) * rstd * n2g[e] + n2b[e];
  }
  __syncthreads();

  // ---- level = sigmoid(16x16x16 reduction with F_w) -> gb, bbv ----
  for (int o = 0; o < 2; o++) {
    float part = 0.f;
    const float* Fo = Fw + o * 4096;
#pragma unroll
    for (int e = 0; e < 16; e++) part += SRCf[tid * 16 + e] * Fo[e * 256 + tid];
    RED[tid] = part;
    __syncthreads();
    for (int st = 128; st > 0; st >>= 1) {
      if (tid < st) RED[tid] += RED[tid + st];
      __syncthreads();
    }
    if (tid == 0) {
      float lvl = 1.f / (1.f + __expf(-(RED[0] + Fb[o])));
      if (o == 0) {
        float g = 0.1f * lvl + 0.18f;
        GB[b] = g;
        if (b == 0) OUT[GOFF] = g;
      } else {
        float bb = 0.04f * lvl + 0.06f;
        BBv[b] = bb;
        if (b == 0) OUT[BOFF] = bb;
      }
    }
    __syncthreads();
  }
}

// ---------------- final per-pixel enhancement ----------------
__global__ void k_final(const float* __restrict__ X,
                        const float* __restrict__ GB, const float* __restrict__ BBv,
                        float* __restrict__ OUT) {
  int idx = blockIdx.x * 256 + threadIdx.x;
  if (idx >= NPIX) return;
  int b = idx >> 20; int rem = idx & (HW - 1);
  size_t xb = (size_t)b * 3 * HW + rem;
  float red = X[xb], green = X[xb + HW], blue = X[xb + 2 * HW];
  float v = (red + green + blue) * (1.f / 3.f);
  float rr = OUT[ROFF + idx];
  float gbv = GB[b], bbv = BBv[b];
  float v0 = fminf(fmaxf(v, 1e-6f), 0.999999f);
  float r0 = __expf(rr * __logf(gbv));
  float ev = __expf(r0 * __logf(v0));
  float d = bbv - v;
  float L = 400.f * d * d * d;
  L = (L < 1e-5f) ? 1e-6f : L;
  ev -= L;
  float s = ev / (v + 1e-6f);
  float er = red * s, eg = green * s, eb = blue * s;
  bool z = v > 0.04f;
  float tr = z ? 0.f : er, tg = z ? 0.f : eg, tb = z ? 0.f : eb;
  OUT[E0OFF + xb] = er; OUT[E0OFF + xb + HW] = eg; OUT[E0OFF + xb + 2 * HW] = eb;
  OUT[E1OFF + xb] = er; OUT[E1OFF + xb + HW] = eg; OUT[E1OFF + xb + 2 * HW] = eb;
  OUT[TOFF + xb]  = tr; OUT[TOFF + xb + HW]  = tg; OUT[TOFF + xb + 2 * HW]  = tb;
}

// ---------------------------------------------------------------------------
extern "C" void kernel_launch(void* const* d_in, const int* in_sizes, int n_in,
                              void* d_out, int out_size, void* d_ws, size_t ws_size,
                              hipStream_t stream) {
  (void)in_sizes; (void)n_in; (void)out_size; (void)ws_size;
  const float* P[47];
  for (int i = 0; i < 47; i++) P[i] = (const float*)d_in[i];
  const float* X = P[0];
  // params in setup_inputs insertion order
  const float *e1w = P[1],  *e1b = P[2],  *pw1w = P[3],  *pw1b = P[4];
  const float *e2w = P[5],  *e2b = P[6],  *pw2w = P[7],  *pw2b = P[8];
  const float *e3w = P[9],  *e3b = P[10], *pw3w = P[11], *pw3b = P[12];
  const float *e4w = P[13], *e4b = P[14], *pw4w = P[15], *pw4b = P[16];
  const float *e5w = P[17], *e5b = P[18], *pw5w = P[19], *pw5b = P[20];
  const float *e6w = P[21], *e6b = P[22], *pw6w = P[23], *pw6b = P[24];
  const float *e7w = P[25], *e7b = P[26], *pw7w = P[27], *pw7b = P[28];
  const float *mw  = P[29], *mb  = P[30], *ing  = P[31], *inb  = P[32];
  const float *aiw = P[33], *aib = P[34], *aow  = P[35], *aob  = P[36];
  const float *l1w = P[37], *l1b = P[38], *l2w  = P[39], *l2b  = P[40];
  const float *n1g = P[41], *n1b = P[42], *n2g  = P[43], *n2b  = P[44];
  const float *Fw  = P[45], *Fb  = P[46];

  // workspace layout: v (f32) | 5 x f16 feature buffers | gb[8] bb[8]
  float* wsf = (float*)d_ws;
  float* V = wsf;
  _Float16* fb0 = (_Float16*)(wsf + NPIX);
  h4f* B1 = (h4f*)(fb0 + (size_t)0 * NPIX * 4);
  h4f* B2 = (h4f*)(fb0 + (size_t)1 * NPIX * 4);
  h4f* B3 = (h4f*)(fb0 + (size_t)2 * NPIX * 4);
  h4f* B4 = (h4f*)(fb0 + (size_t)3 * NPIX * 4);
  h4f* B5 = (h4f*)(fb0 + (size_t)4 * NPIX * 4);
  float* GB  = (float*)(fb0 + (size_t)5 * NPIX * 4);
  float* BBv = GB + 8;

  float* OUT = (float*)d_out;
  dim3 blk(256), grd(NPIX / 256);

  k_lum   <<<grd, blk, 0, stream>>>(X, V);
  k_stage1<<<grd, blk, 0, stream>>>(V, B1, e1w, e1b, pw1w, pw1b);
  k_dwpw4 <<<grd, blk, 0, stream>>>(B1, B2, e2w, e2b, pw2w, pw2b);
  k_dwpw4 <<<grd, blk, 0, stream>>>(B2, B3, e3w, e3b, pw3w, pw3b);
  k_dwpw4 <<<grd, blk, 0, stream>>>(B3, B4, e4w, e4b, pw4w, pw4b);
  k_dwpw8 <<<grd, blk, 0, stream>>>(B3, B4, B5, e5w, e5b, pw5w, pw5b);
  k_dwpw8 <<<grd, blk, 0, stream>>>(B2, B5, B3, e6w, e6b, pw6w, pw6b);  // p6 -> B3
  k_stage7<<<grd, blk, 0, stream>>>(B1, B3, OUT + ROFF, e7w, e7b, pw7w, pw7b);

  // LDS bytes: f32 region 25888*4 + f16 region 57344*2 = 218240 (< 320KB WGP)
  size_t smem = (size_t)25888 * sizeof(float) + (size_t)57344 * sizeof(_Float16);
  k_small<<<dim3(8), blk, smem, stream>>>(V, mw, mb, ing, inb,
                                          aiw, aib, aow, aob,
                                          l1w, l1b, l2w, l2b,
                                          n1g, n1b, n2g, n2b,
                                          Fw, Fb, GB, BBv, OUT);

  k_final<<<grd, blk, 0, stream>>>(X, GB, BBv, OUT);
}
